// CBSA_57114475102375
// MI455X (gfx1250) — matrix operand order repaired and checked by reading, not compiled, not run on hardware
//
#include <hip/hip_runtime.h>

typedef __attribute__((ext_vector_type(16))) __bf16 v16bf;
typedef __attribute__((ext_vector_type(8)))  __bf16 v8bf;
typedef __attribute__((ext_vector_type(8)))  float  v8f;

#define DEV static __device__ __forceinline__

static constexpr int B_    = 128;
static constexpr int N_    = 577;
static constexpr int DIM_  = 1024;
static constexpr int HEADS_= 16;
static constexpr int DH_   = 64;
static constexpr int M_    = B_ * N_;      // 73856 rows for the big GEMMs (577 * 128)
static constexpr int NPAD  = 592;          // 37*16 : token dim padded for 16-wide N tiles
static constexpr int NPAD2 = 608;          // 19*32 : token dim padded for 32-wide K steps

DEV v8f wmma_bf16(v16bf a, v16bf b, v8f c) {
  return __builtin_amdgcn_wmma_f32_16x16x32_bf16(false, a, false, b, (short)0, c,
                                                 false, false);
}

DEV v16bf cat8(v8bf lo, v8bf hi) {
  v16bf r;
#pragma unroll
  for (int i = 0; i < 8; ++i) { r[i] = lo[i]; r[i + 8] = hi[i]; }
  return r;
}

// A fragment (16x32) from an f32 row-major matrix, converting to bf16 in-flight.
DEV v16bf load_a_f32(const float* base, int ldm, int m0, int k0, int lane) {
  int m  = m0 + (lane & 15);
  int hi = lane >> 4;
  const float* p0 = base + (size_t)m * ldm + k0 + hi * 8;
  v16bf a;
#pragma unroll
  for (int i = 0; i < 8; ++i) {
    a[i]     = (__bf16)p0[i];
    a[i + 8] = (__bf16)p0[16 + i];
  }
  return a;
}

// A fragment (16x32) from a bf16 row-major matrix.
DEV v16bf load_a_bf16(const __bf16* base, int ldm, int m0, int k0, int lane) {
  int m  = m0 + (lane & 15);
  int hi = lane >> 4;
  const __bf16* p0 = base + (size_t)m * ldm + k0 + hi * 8;
  v8bf lo = *(const v8bf*)p0;
  v8bf h8 = *(const v8bf*)(p0 + 16);
  return cat8(lo, h8);
}

// B fragment (32x16) from a bf16 [K][N] row-major matrix.
DEV v16bf load_b_bf16(const __bf16* base, int ldn, int k0, int n0, int lane) {
  int hi = lane >> 4;
  const __bf16* p = base + (size_t)(k0 + hi * 8) * ldn + n0 + (lane & 15);
  v16bf b;
#pragma unroll
  for (int i = 0; i < 8; ++i) {
    b[i]     = p[(size_t)i * ldn];
    b[i + 8] = p[(size_t)(16 + i) * ldn];
  }
  return b;
}

// Async global->LDS 16B transfer (CDNA5 VGLOBAL async path, ASYNCcnt-tracked).
DEV void async_g2l_b128(unsigned lds_byte_addr, const void* gaddr) {
  asm volatile("global_load_async_to_lds_b128 %0, %1, off"
               :: "v"(lds_byte_addr), "v"(gaddr) : "memory");
}
DEV void wait_async0() { asm volatile("s_wait_asynccnt 0" ::: "memory"); }

DEV unsigned lds_addr_of(const void* p) {
  return (unsigned)(unsigned long long)p;   // generic LDS addr truncates to LDS offset
}

// ---------------------------------------------------------------------------
// K0: transpose + convert both weight matrices to bf16 [K][N] layout.
__global__ __launch_bounds__(256) void k_cvtW(const float* __restrict__ Wp,
                                              const float* __restrict__ Wo,
                                              __bf16* __restrict__ WpT,
                                              __bf16* __restrict__ WoT) {
  int idx = blockIdx.x * 256 + threadIdx.x;   // 1M elements
  int i = idx >> 10, k = idx & 1023;
  WpT[(size_t)k * 1024 + i] = (__bf16)Wp[idx];
  WoT[(size_t)k * 1024 + i] = (__bf16)Wo[idx];
}

// ---------------------------------------------------------------------------
// K1: w = x @ W_proj^T  (M=73856, N=1024, K=1024), output bf16.
// 577 workgroups x 128-row M strip; 8 N-blocks of 128. B panel (32x128, 8KB)
// ping-pong double-buffered via async global->LDS: panel kt+1 is prefetched
// while the 8 WMMAs of panel kt execute; one barrier per k-step.
__global__ __launch_bounds__(256) void k_proj_gemm(const float* __restrict__ x,
                                                   const __bf16* __restrict__ WpT,
                                                   __bf16* __restrict__ w) {
  __shared__ __bf16 sB[2][32][128];   // 2 x 8 KB
  int m0   = blockIdx.x * 128;
  int tid  = threadIdx.x;
  int lane = tid & 31;
  int wv   = tid >> 5;
  int mw   = m0 + wv * 16;
  int hi   = lane >> 4, nn = lane & 15;
  unsigned ldsB = lds_addr_of(&sB[0][0][0]);
  int f    = tid * 32;              // this thread's 32-byte slice of a panel
  int prow = f >> 8;                // panel row (k within step)
  int pcol = (f & 255) >> 1;        // element col within row

  for (int nb = 0; nb < 8; ++nb) {
    int nbase = nb * 128;
    v8f acc[8];
#pragma unroll
    for (int j = 0; j < 8; ++j) acc[j] = v8f{};

    // prefetch panel 0 into buffer 0
    {
      const __bf16* g = WpT + (size_t)prow * DIM_ + nbase + pcol;
      async_g2l_b128(ldsB + f, g);
      async_g2l_b128(ldsB + f + 16, g + 8);
    }
    wait_async0();
    __syncthreads();

    for (int kt = 0; kt < 32; ++kt) {
      int k0  = kt * 32;
      int cur = kt & 1;
      if (kt < 31) {   // prefetch next panel into the other buffer
        const __bf16* g = WpT + (size_t)(k0 + 32 + prow) * DIM_ + nbase + pcol;
        unsigned dst = ldsB + (cur ^ 1) * 8192 + f;
        async_g2l_b128(dst, g);
        async_g2l_b128(dst + 16, g + 8);
      }
      v16bf a = load_a_f32(x, DIM_, mw, k0, lane);
#pragma unroll
      for (int j = 0; j < 8; ++j) {
        v16bf b = load_b_bf16(&sB[cur][0][0], 128, 0, j * 16, lane);
        acc[j] = wmma_bf16(a, b, acc[j]);
      }
      wait_async0();     // next panel has landed in LDS
      __syncthreads();   // all waves done reading current panel
    }
#pragma unroll
    for (int j = 0; j < 8; ++j) {
#pragma unroll
      for (int r = 0; r < 8; ++r) {
        int m = mw + r + 8 * hi;
        int n = nbase + j * 16 + nn;
        w[(size_t)m * DIM_ + n] = (__bf16)acc[j][r];
      }
    }
  }
}

// ---------------------------------------------------------------------------
// K2: adaptive 8x8 pool (3x3 block mean on 24x24 grid) -> rep0 f32 (b,h,64,64)
__global__ __launch_bounds__(256) void k_pool(const __bf16* __restrict__ w,
                                              float* __restrict__ rep) {
  int idx = blockIdx.x * 256 + threadIdx.x;   // 128*16*64*64 = 8388608
  int d = idx & 63, p = (idx >> 6) & 63, h = (idx >> 12) & 15, b = idx >> 16;
  int py = p >> 3, px = p & 7;
  const __bf16* base = w + ((size_t)b * N_) * DIM_ + h * DH_ + d;
  float s = 0.f;
#pragma unroll
  for (int i = 0; i < 3; ++i)
#pragma unroll
    for (int j = 0; j < 3; ++j) {
      int tok = (py * 3 + i) * 24 + (px * 3 + j);
      s += (float)base[(size_t)tok * DIM_];
    }
  rep[idx] = s * (1.0f / 9.0f);
}

// ---------------------------------------------------------------------------
// K3: attention 1 per (b,h). S = rep @ wq^T (64x577), softmax rows.
// Stores unnormalized P transposed [tok][m] (bf16), 1/rowsum (Linv),
// updated rep = rep + step_rep * (P@v)/rowsum (f32 + bf16 + bf16-transposed).
__global__ __launch_bounds__(256) void k_attn1(const __bf16* __restrict__ w,
                                               float* __restrict__ rep,
                                               const float* __restrict__ step_rep,
                                               __bf16* __restrict__ PT,
                                               __bf16* __restrict__ repb,
                                               __bf16* __restrict__ rept,
                                               float* __restrict__ Linv) {
  __shared__ float  sS[64][NPAD2];   // 155648 B
  __shared__ __bf16 sKT[64][NPAD];   //  75776 B  (headdim x token)
  __shared__ float  sSum[64];

  int bh = blockIdx.x, b = bh >> 4, h = bh & 15;
  int tid = threadIdx.x, lane = tid & 31, wv = tid >> 5;
  const __bf16* wq = w + ((size_t)b * N_) * DIM_ + h * DH_;   // 577 rows, ld=1024
  float* q = rep + (size_t)bh * 4096;                          // 64x64 f32

  // stage K^T into LDS (zero-pad tokens >= 577)
  for (int i = tid; i < NPAD * 64; i += 256) {
    int tok = i >> 6, kd = i & 63;
    sKT[kd][tok] = (tok < N_) ? wq[(size_t)tok * DIM_ + kd] : (__bf16)0.0f;
  }
  __syncthreads();

  // S = q @ K^T, scaled by 1/8. 4 M-tiles x 37 N-tiles = 148 C tiles.
  const float scale = 0.125f;
  for (int t = wv; t < 148; t += 8) {
    int mt = t & 3, nt = t >> 2;
    v8f c = v8f{};
#pragma unroll
    for (int kt = 0; kt < 2; ++kt) {
      v16bf a  = load_a_f32(q, DH_, mt * 16, kt * 32, lane);
      v16bf bm = load_b_bf16(&sKT[0][0], NPAD, kt * 32, nt * 16, lane);
      c = wmma_bf16(a, bm, c);
    }
    int hi = lane >> 4, nn = lane & 15;
#pragma unroll
    for (int r = 0; r < 8; ++r)
      sS[mt * 16 + r + 8 * hi][nt * 16 + nn] = c[r] * scale;
  }
  __syncthreads();

  // row softmax (unnormalized exp kept in sS, sum saved)
  if (tid < 64) {
    float mx = -1e30f;
    for (int tk = 0; tk < N_; ++tk) mx = fmaxf(mx, sS[tid][tk]);
    float s = 0.f;
    for (int tk = 0; tk < N_; ++tk) {
      float e = __expf(sS[tid][tk] - mx);
      sS[tid][tk] = e;
      s += e;
    }
    for (int tk = N_; tk < NPAD2; ++tk) sS[tid][tk] = 0.f;
    sSum[tid] = s;
    Linv[(size_t)bh * 64 + tid] = 1.0f / s;
  }
  __syncthreads();

  // persist P transposed [tok][m] as bf16 for the later attn^T GEMM
  __bf16* pt = PT + (size_t)bh * NPAD * 64;
  for (int i = tid; i < NPAD * 64; i += 256) {
    int tok = i >> 6, m = i & 63;
    pt[i] = (__bf16)sS[m][tok];
  }

  // O = P @ V : 16 C tiles / 8 waves -> 2 per wave. K = 608 (19 steps).
  int mt  = wv >> 1;
  int ntb = (wv & 1) * 2;
  v8f o0 = v8f{}, o1 = v8f{};
  for (int kt = 0; kt < 19; ++kt) {
    v16bf a = load_a_f32(&sS[0][0], NPAD2, mt * 16, kt * 32, lane);
    v16bf b0 = load_b_bf16(wq, DIM_, kt * 32, ntb * 16, lane);
    v16bf b1 = load_b_bf16(wq, DIM_, kt * 32, ntb * 16 + 16, lane);
    o0 = wmma_bf16(a, b0, o0);
    o1 = wmma_bf16(a, b1, o1);
  }

  // rep_new = rep + step_rep[h] * O / rowsum ; write f32, bf16, bf16^T
  float sr = step_rep[h];
  int hi = lane >> 4, nn = lane & 15;
  __bf16* rb = repb + (size_t)bh * 4096;
  __bf16* rt = rept + (size_t)bh * 4096;
#pragma unroll
  for (int r = 0; r < 8; ++r) {
    int m = mt * 16 + r + 8 * hi;
    float inv = 1.0f / sSum[m];
    {
      int n = ntb * 16 + nn;
      float v = q[m * DH_ + n] + sr * o0[r] * inv;
      q[m * DH_ + n] = v;
      rb[m * DH_ + n] = (__bf16)v;
      rt[n * DH_ + m] = (__bf16)v;
    }
    {
      int n = ntb * 16 + 16 + nn;
      float v = q[m * DH_ + n] + sr * o1[r] * inv;
      q[m * DH_ + n] = v;
      rb[m * DH_ + n] = (__bf16)v;
      rt[n * DH_ + m] = (__bf16)v;
    }
  }
}

// ---------------------------------------------------------------------------
// K4: 64x64 self-attention on rep per (b,h); output
// xd2s[m][d] = step_x[h] * (softmax(rep rep^T /8) @ rep)[m][d] * Linv1[m]  (bf16)
__global__ __launch_bounds__(256) void k_attn2(const __bf16* __restrict__ repb,
                                               const __bf16* __restrict__ rept,
                                               const float* __restrict__ Linv,
                                               const float* __restrict__ step_x,
                                               __bf16* __restrict__ xd2) {
  __shared__ float  sS[64][64];
  __shared__ __bf16 sP[64][64];
  __shared__ float  sSum[64];

  int bh = blockIdx.x, h = bh & 15;
  int tid = threadIdx.x, lane = tid & 31, wv = tid >> 5;
  const __bf16* A  = repb + (size_t)bh * 4096;   // rep row-major
  const __bf16* Bt = rept + (size_t)bh * 4096;   // rep^T row-major
  int mt = wv >> 1, ntb = (wv & 1) * 2;
  int hi = lane >> 4, nn = lane & 15;

  v8f c0 = v8f{}, c1 = v8f{};
#pragma unroll
  for (int kt = 0; kt < 2; ++kt) {
    v16bf a  = load_a_bf16(A, DH_, mt * 16, kt * 32, lane);
    v16bf b0 = load_b_bf16(Bt, DH_, kt * 32, ntb * 16, lane);
    v16bf b1 = load_b_bf16(Bt, DH_, kt * 32, ntb * 16 + 16, lane);
    c0 = wmma_bf16(a, b0, c0);
    c1 = wmma_bf16(a, b1, c1);
  }
#pragma unroll
  for (int r = 0; r < 8; ++r) {
    sS[mt * 16 + r + 8 * hi][ntb * 16 + nn]      = c0[r] * 0.125f;
    sS[mt * 16 + r + 8 * hi][ntb * 16 + 16 + nn] = c1[r] * 0.125f;
  }
  __syncthreads();

  if (tid < 64) {
    float mx = -1e30f;
    for (int c = 0; c < 64; ++c) mx = fmaxf(mx, sS[tid][c]);
    float s = 0.f;
    for (int c = 0; c < 64; ++c) {
      float e = __expf(sS[tid][c] - mx);
      sP[tid][c] = (__bf16)e;
      s += e;
    }
    sSum[tid] = s;
  }
  __syncthreads();

  v8f o0 = v8f{}, o1 = v8f{};
#pragma unroll
  for (int kt = 0; kt < 2; ++kt) {
    v16bf a  = load_a_bf16(&sP[0][0], DH_, mt * 16, kt * 32, lane);
    v16bf b0 = load_b_bf16(A, DH_, kt * 32, ntb * 16, lane);
    v16bf b1 = load_b_bf16(A, DH_, kt * 32, ntb * 16 + 16, lane);
    o0 = wmma_bf16(a, b0, o0);
    o1 = wmma_bf16(a, b1, o1);
  }

  float sx = step_x[h];
  __bf16* out = xd2 + (size_t)bh * 4096;
#pragma unroll
  for (int r = 0; r < 8; ++r) {
    int m = mt * 16 + r + 8 * hi;
    float fscale = sx / sSum[m] * Linv[(size_t)bh * 64 + m];
    out[m * DH_ + ntb * 16 + nn]      = (__bf16)(o0[r] * fscale);
    out[m * DH_ + ntb * 16 + 16 + nn] = (__bf16)(o1[r] * fscale);
  }
}

// ---------------------------------------------------------------------------
// K5: xd[tok][d] = sum_m P[m][tok] * xd2s[m][d]  per (b,h)  (attn^T GEMM)
__global__ __launch_bounds__(256) void k_xd(const __bf16* __restrict__ PT,
                                            const __bf16* __restrict__ xd2,
                                            __bf16* __restrict__ xd) {
  int bh = blockIdx.x, b = bh >> 4, h = bh & 15;
  int lane = threadIdx.x & 31, wv = threadIdx.x >> 5;
  const __bf16* A  = PT  + (size_t)bh * NPAD * 64;  // [tok][m] row-major
  const __bf16* Bm = xd2 + (size_t)bh * 4096;       // [m][d] row-major
  __bf16* out = xd + ((size_t)b * N_) * DIM_ + h * DH_;

  for (int t = wv; t < 148; t += 8) {   // 37 tok-tiles x 4 d-tiles
    int mt = t >> 2, nt = t & 3;
    v8f c = v8f{};
#pragma unroll
    for (int kt = 0; kt < 2; ++kt) {
      v16bf a  = load_a_bf16(A, DH_, mt * 16, kt * 32, lane);
      v16bf bm = load_b_bf16(Bm, DH_, kt * 32, nt * 16, lane);
      c = wmma_bf16(a, bm, c);
    }
    int hi = lane >> 4, nn = lane & 15;
#pragma unroll
    for (int r = 0; r < 8; ++r) {
      int tok = mt * 16 + r + 8 * hi;
      if (tok < N_) out[(size_t)tok * DIM_ + nt * 16 + nn] = (__bf16)c[r];
    }
  }
}

// ---------------------------------------------------------------------------
// K6: out = xd @ W_out^T + b_out  (M=73856, N=1024, K=1024), f32 output.
// Same double-buffered async B panel as K1.
__global__ __launch_bounds__(256) void k_out_gemm(const __bf16* __restrict__ xd,
                                                  const __bf16* __restrict__ WoT,
                                                  const float* __restrict__ bias,
                                                  float* __restrict__ out) {
  __shared__ __bf16 sB[2][32][128];   // 2 x 8 KB
  int m0   = blockIdx.x * 128;
  int tid  = threadIdx.x;
  int lane = tid & 31;
  int wv   = tid >> 5;
  int mw   = m0 + wv * 16;
  int hi   = lane >> 4, nn = lane & 15;
  unsigned ldsB = lds_addr_of(&sB[0][0][0]);
  int f    = tid * 32;
  int prow = f >> 8;
  int pcol = (f & 255) >> 1;

  for (int nb = 0; nb < 8; ++nb) {
    int nbase = nb * 128;
    v8f acc[8];
#pragma unroll
    for (int j = 0; j < 8; ++j) acc[j] = v8f{};

    {
      const __bf16* g = WoT + (size_t)prow * DIM_ + nbase + pcol;
      async_g2l_b128(ldsB + f, g);
      async_g2l_b128(ldsB + f + 16, g + 8);
    }
    wait_async0();
    __syncthreads();

    for (int kt = 0; kt < 32; ++kt) {
      int k0  = kt * 32;
      int cur = kt & 1;
      if (kt < 31) {
        const __bf16* g = WoT + (size_t)(k0 + 32 + prow) * DIM_ + nbase + pcol;
        unsigned dst = ldsB + (cur ^ 1) * 8192 + f;
        async_g2l_b128(dst, g);
        async_g2l_b128(dst + 16, g + 8);
      }
      v16bf a = load_a_bf16(xd, DIM_, mw, k0, lane);
#pragma unroll
      for (int j = 0; j < 8; ++j) {
        v16bf b = load_b_bf16(&sB[cur][0][0], 128, 0, j * 16, lane);
        acc[j] = wmma_bf16(a, b, acc[j]);
      }
      wait_async0();
      __syncthreads();
    }
#pragma unroll
    for (int j = 0; j < 8; ++j) {
#pragma unroll
      for (int r = 0; r < 8; ++r) {
        int m = mw + r + 8 * hi;
        int n = nbase + j * 16 + nn;
        out[(size_t)m * DIM_ + n] = acc[j][r] + bias[n];
      }
    }
  }
}

// ---------------------------------------------------------------------------
extern "C" void kernel_launch(void* const* d_in, const int* in_sizes, int n_in,
                              void* d_out, int out_size, void* d_ws, size_t ws_size,
                              hipStream_t stream) {
  (void)in_sizes; (void)n_in; (void)out_size; (void)ws_size;
  const float* x        = (const float*)d_in[0];
  const float* W_proj   = (const float*)d_in[1];
  const float* step_x   = (const float*)d_in[2];
  const float* step_rep = (const float*)d_in[3];
  const float* W_out    = (const float*)d_in[4];
  const float* b_out    = (const float*)d_in[5];
  float* out = (float*)d_out;

  char* ws = (char*)d_ws;
  size_t off = 0;
  __bf16* WpT  = (__bf16*)(ws + off); off += (size_t)1024 * 1024 * 2;
  __bf16* WoT  = (__bf16*)(ws + off); off += (size_t)1024 * 1024 * 2;
  __bf16* w    = (__bf16*)(ws + off); off += (size_t)M_ * DIM_ * 2;
  __bf16* PT   = (__bf16*)(ws + off); off += (size_t)2048 * NPAD * 64 * 2;
  float*  rep  = (float*)(ws + off);  off += (size_t)2048 * 4096 * 4;
  __bf16* repb = (__bf16*)(ws + off); off += (size_t)2048 * 4096 * 2;
  __bf16* rept = (__bf16*)(ws + off); off += (size_t)2048 * 4096 * 2;
  float*  Linv = (float*)(ws + off);  off += (size_t)2048 * 64 * 4;
  __bf16* xd2  = (__bf16*)(ws + off); off += (size_t)2048 * 4096 * 2;
  __bf16* xd   = (__bf16*)(ws + off); off += (size_t)M_ * DIM_ * 2;

  k_cvtW    <<<4096, 256, 0, stream>>>(W_proj, W_out, WpT, WoT);
  k_proj_gemm<<<577, 256, 0, stream>>>(x, WpT, w);
  k_pool    <<<32768, 256, 0, stream>>>(w, rep);
  k_attn1   <<<2048, 256, 0, stream>>>(w, rep, step_rep, PT, repb, rept, Linv);
  k_attn2   <<<2048, 256, 0, stream>>>(repb, rept, Linv, step_x, xd2);
  k_xd      <<<2048, 256, 0, stream>>>(PT, xd2, xd);
  k_out_gemm<<<577, 256, 0, stream>>>(xd, WoT, b_out, out);
}
